// RC_37529424233355
// MI455X (gfx1250) — matrix operand order repaired and checked by reading, not compiled, not run on hardware
//
#include <hip/hip_runtime.h>
#include <hip/hip_bf16.h>
#include <math.h>

// ---------------------------------------------------------------------------
// Adaptive spiking RNN forward, CDNA5 (gfx1250), wave32 + WMMA bf16.
//
// B=256, T=250, I=700(->704), H=512, O=20(->32).
// Grid: 16 blocks (one per 16 batch rows), 512 threads = 16 wave32.
// Each wave owns 32 hidden columns (2 WMMA 16x16 N-tiles) of BOTH layers;
// mem/spk/spike-count state lives in VGPRs in WMMA C-layout; spikes are
// published to LDS (bf16) each step for the next GEMM's A-fragments.
// Weights: bf16 copies in d_ws (masked recurrents), resident in L2; an
// opaque per-step offset stops LICM from hoisting (and spilling) them.
// ---------------------------------------------------------------------------

typedef __attribute__((ext_vector_type(16))) __bf16 v16bf;
typedef __attribute__((ext_vector_type(8)))  __bf16 v8bf;
typedef __attribute__((ext_vector_type(8)))  float  v8f;

#define SB_J0 0.01f
#define SBETA 1.8f

// A-fragment (16x32 bf16, row-major MxK, stride ld elements).
// lanes 0-15: row=lane, K chunks [k0..k0+7] and [k0+16..k0+23]
// lanes16-31: row=lane-16, K chunks [k0+8..k0+15] and [k0+24..k0+31]
__device__ __forceinline__ v16bf load_fragA(const __bf16* __restrict__ base,
                                            int ld, int k0, int lane) {
  const int g = (lane >> 4) << 3;            // 0 or 8
  const int r = lane & 15;
  const __bf16* p = base + r * ld + k0 + g;
  union { v16bf v; v8bf h[2]; } f;
  f.h[0] = *(const v8bf*)(p);
  f.h[1] = *(const v8bf*)(p + 16);
  return f.v;
}

// B-fragment (32x16 bf16, K x N) from a weight stored row-major [N,K]:
// lane n (0-15) holds col n, K=[k0..k0+7],[k0+16..k0+23]; lane n+16 holds K=[k0+8..k0+15],[k0+24..k0+31].
__device__ __forceinline__ v16bf load_fragB(const __bf16* __restrict__ base,
                                            int ld, int k0, int lane) {
  const int r = lane & 15;
  const __bf16* p = base + r * ld + k0 + ((lane >> 4) << 3);
  union { v16bf v; v8bf h[2]; } f;
  f.h[0] = *(const v8bf*)(p);
  f.h[1] = *(const v8bf*)(p + 16);
  return f.v;
}

// GEMM over one shared A (row-major in LDS) and two adjacent B N-tiles.
template <int KMAX, int LDA, int LDB>
__device__ __forceinline__ void gemm2(const __bf16* __restrict__ A,
                                      const __bf16* __restrict__ B0,
                                      int lane, v8f& c0, v8f& c1) {
  const __bf16* B1 = B0 + 16 * LDB;
  for (int k0 = 0; k0 < KMAX; k0 += 32) {
    v16bf a  = load_fragA(A, LDA, k0, lane);
    v16bf b0 = load_fragB(B0, LDB, k0, lane);
    v16bf b1 = load_fragB(B1, LDB, k0, lane);
    c0 = __builtin_amdgcn_wmma_f32_16x16x32_bf16(false, a, false, b0, (short)0, c0, false, false);
    c1 = __builtin_amdgcn_wmma_f32_16x16x32_bf16(false, a, false, b1, (short)0, c1, false, false);
  }
}

// Adaptive-threshold LIF neuron update on one 16x16 C-tile (per-lane column).
__device__ __forceinline__ void neuron_update(v8f c, float bias, float alpha,
                                              float ro, v8f& mem, v8f& sp, v8f& s) {
  v8f so = sp;
  v8f b  = ro * SB_J0 + (1.f - ro) * so;
  v8f Bt = SB_J0 + SBETA * b;
  v8f cur = c + bias;
  mem = mem * alpha + (1.f - alpha) * cur - Bt * so;
  v8f sn;
#pragma unroll
  for (int k = 0; k < 8; ++k) sn[k] = (mem[k] - Bt[k]) > 0.f ? 1.f : 0.f;
  sp = sn;
  s += sn;
}

// ------------------------- weight pre-pass kernels -------------------------

__global__ void prep_zero(float* anorm) { *anorm = 0.f; }

__global__ void prep_win(const float* __restrict__ W, __bf16* __restrict__ Wb) {
  int idx = blockIdx.x * blockDim.x + threadIdx.x;      // over 512*704
  if (idx >= 512 * 704) return;
  int n = idx / 704, c = idx - n * 704;
  float v = (c < 700) ? W[n * 700 + c] : 0.f;
  Wb[idx] = (__bf16)v;
}

// Square 512x512 weight; optional mask-plane (W * mask.T) + |.| reduction.
__global__ void prep_sq(const float* __restrict__ W, const float* __restrict__ mask,
                        __bf16* __restrict__ Wb, float* __restrict__ anorm) {
  int idx = blockIdx.x * blockDim.x + threadIdx.x;
  float v = 0.f;
  if (idx < 512 * 512) {
    int n = idx >> 9, k = idx & 511;
    v = W[idx];
    if (mask) v *= mask[k * 512 + n];                   // mask.T
    Wb[idx] = (__bf16)v;
  }
  if (mask) {
    __shared__ float red[256];
    red[threadIdx.x] = fabsf(v);
    __syncthreads();
    for (int s = 128; s > 0; s >>= 1) {
      if (threadIdx.x < s) red[threadIdx.x] += red[threadIdx.x + s];
      __syncthreads();
    }
    if (threadIdx.x == 0) atomicAdd(anorm, red[0]);
  }
}

__global__ void prep_wout(const float* __restrict__ W, __bf16* __restrict__ Wb) {
  int idx = blockIdx.x * blockDim.x + threadIdx.x;      // over 32*512
  if (idx >= 32 * 512) return;
  int n = idx >> 9, k = idx & 511;
  Wb[idx] = (__bf16)((n < 20) ? W[n * 512 + k] : 0.f);
}

// ------------------------------ main kernel --------------------------------

__global__ __launch_bounds__(512, 1)
void srnn_kernel(const float* __restrict__ input,
                 const float* __restrict__ h1m0, const float* __restrict__ h2m0,
                 const float* __restrict__ om0,
                 const float* __restrict__ b_in, const float* __restrict__ b_h11,
                 const float* __restrict__ b_h12, const float* __restrict__ b_h22,
                 const float* __restrict__ b_o,
                 const float* __restrict__ tau_adp_h1, const float* __restrict__ tau_adp_h2,
                 const float* __restrict__ tau_m_h1, const float* __restrict__ tau_m_h2,
                 const float* __restrict__ tau_m_o,
                 const __bf16* __restrict__ Wb_in,   // [512][704]
                 const __bf16* __restrict__ Wb_h11,  // [512][512] masked
                 const __bf16* __restrict__ Wb_h12,  // [512][512]
                 const __bf16* __restrict__ Wb_h22,  // [512][512] masked
                 const __bf16* __restrict__ Wb_out,  // [32][512]  zero-padded
                 float* __restrict__ out_acc,        // [256][20]
                 float* __restrict__ out_s1,         // [256][512]
                 float* __restrict__ out_s2) {       // [256][512]
  const int tid  = threadIdx.x;
  const int lane = tid & 31;
  const int w    = tid >> 5;              // wave 0..15, owns H cols [32w,32w+32)
  const int rb   = blockIdx.x * 16;       // batch row base
  const int g8   = (lane >> 4) * 8;       // C-layout row offset
  const int cn   = lane & 15;             // C-layout col-in-tile

  __shared__ __attribute__((aligned(16))) __bf16 xt[16 * 704];
  __shared__ __attribute__((aligned(16))) __bf16 sp1s[16 * 512];
  __shared__ __attribute__((aligned(16))) __bf16 sp2s[16 * 512];
  __shared__ float ios[16 * 32];
  __shared__ float outms[16 * 20];
  __shared__ float accs[16 * 20];
  __shared__ float alphos[32];

  float alpha1[2], ro1[2], bias1[2], alpha2[2], ro2[2], bias2[2];
  const int nb0 = w * 32;
  v8f mem1[2], mem2[2], s1[2], s2[2], sp1[2], sp2[2];

  for (int j = 0; j < 2; ++j) {
    const int n = nb0 + j * 16 + cn;
    alpha1[j] = __expf(-1.0f / tau_m_h1[n]);
    ro1[j]    = __expf(-1.0f / tau_adp_h1[n]);
    bias1[j]  = b_in[n] + b_h11[n];
    alpha2[j] = __expf(-1.0f / tau_m_h2[n]);
    ro2[j]    = __expf(-1.0f / tau_adp_h2[n]);
    bias2[j]  = b_h12[n] + b_h22[n];
    for (int i = 0; i < 8; ++i) {
      const int m = rb + g8 + i;
      mem1[j][i] = h1m0[m * 512 + n];
      mem2[j][i] = h2m0[m * 512 + n];
      s1[j][i] = 0.f; s2[j][i] = 0.f; sp1[j][i] = 0.f; sp2[j][i] = 0.f;
    }
  }
  for (int idx = tid; idx < 16 * 512; idx += 512) {
    sp1s[idx] = (__bf16)0.f; sp2s[idx] = (__bf16)0.f;
  }
  if (tid < 16 * 20) { outms[tid] = om0[rb * 20 + tid]; accs[tid] = 0.f; }
  if (tid < 20) alphos[tid] = __expf(-1.0f / tau_m_o[tid]);
  __syncthreads();

#pragma unroll 1
  for (int t = 0; t < 250; ++t) {
    // Opaque zero offset: stops LICM from hoisting the (loop-invariant)
    // weight loads out of the t-loop and spilling fragments to scratch.
    int z = 0;
    asm volatile("" : "+s"(z));
    const __bf16* pWin = Wb_in  + z;
    const __bf16* pW11 = Wb_h11 + z;
    const __bf16* pW12 = Wb_h12 + z;
    const __bf16* pW22 = Wb_h22 + z;
    const __bf16* pWo  = Wb_out + z;

    // stage x_t -> LDS (bf16, padded 700->704); rows are 16B aligned,
    // 700 = 175 float4 slots, slot 175 is the zero pad.
    for (int idx = tid; idx < 16 * 176; idx += 512) {
      const int m = idx / 176, c4 = (idx - m * 176) * 4;
      float4 v = make_float4(0.f, 0.f, 0.f, 0.f);
      if (c4 < 700)
        v = *(const float4*)(input + (size_t)(rb + m) * (250 * 700) + (size_t)t * 700 + c4);
      __bf16* dst = &xt[m * 704 + c4];
      dst[0] = (__bf16)v.x; dst[1] = (__bf16)v.y;
      dst[2] = (__bf16)v.z; dst[3] = (__bf16)v.w;
    }
    __syncthreads();

    // ---- layer 1: i1 = x @ Win^T + spk1_old @ Wh11e^T (+bias) ----
    {
      v8f c0 = {}, c1 = {};
      gemm2<704, 704, 704>(xt,   pWin + nb0 * 704, lane, c0, c1);
      gemm2<512, 512, 512>(sp1s, pW11 + nb0 * 512, lane, c0, c1);
      neuron_update(c0, bias1[0], alpha1[0], ro1[0], mem1[0], sp1[0], s1[0]);
      neuron_update(c1, bias1[1], alpha1[1], ro1[1], mem1[1], sp1[1], s1[1]);
    }
    __syncthreads();                       // all waves done reading old sp1s
    for (int j = 0; j < 2; ++j)
      for (int i = 0; i < 8; ++i)
        sp1s[(g8 + i) * 512 + nb0 + j * 16 + cn] = (__bf16)sp1[j][i];
    __syncthreads();                       // new sp1 visible

    // ---- layer 2: i2 = spk1_new @ Wh12^T + spk2_old @ Wh22e^T (+bias) ----
    {
      v8f c0 = {}, c1 = {};
      gemm2<512, 512, 512>(sp1s, pW12 + nb0 * 512, lane, c0, c1);
      gemm2<512, 512, 512>(sp2s, pW22 + nb0 * 512, lane, c0, c1);
      neuron_update(c0, bias2[0], alpha2[0], ro2[0], mem2[0], sp2[0], s2[0]);
      neuron_update(c1, bias2[1], alpha2[1], ro2[1], mem2[1], sp2[1], s2[1]);
    }
    __syncthreads();                       // all waves done reading old sp2s
    for (int j = 0; j < 2; ++j)
      for (int i = 0; i < 8; ++i)
        sp2s[(g8 + i) * 512 + nb0 + j * 16 + cn] = (__bf16)sp2[j][i];
    __syncthreads();                       // new sp2 visible

    // ---- readout: io = spk2 @ Wout^T, leaky integrate + softmax accum ----
    if (w == 0) {                          // wave-uniform branch: EXEC all-1s
      v8f c0 = {}, c1 = {};
      gemm2<512, 512, 512>(sp2s, pWo, lane, c0, c1);
      for (int i = 0; i < 8; ++i) {
        ios[(g8 + i) * 32 + cn]      = c0[i];
        ios[(g8 + i) * 32 + 16 + cn] = c1[i];
      }
    }
    __syncthreads();
    if (tid < 16) {                        // one thread per batch row
      float om[20], mx = -1e30f;
      for (int o = 0; o < 20; ++o) {
        float io = ios[tid * 32 + o] + b_o[o];
        float a  = alphos[o];
        float v  = outms[tid * 20 + o] * a + (1.f - a) * io;
        outms[tid * 20 + o] = v;
        om[o] = v;
        mx = fmaxf(mx, v);
      }
      float sum = 0.f;
      for (int o = 0; o < 20; ++o) { om[o] = __expf(om[o] - mx); sum += om[o]; }
      float r = 1.f / sum;
      for (int o = 0; o < 20; ++o) accs[tid * 20 + o] += om[o] * r;
    }
    __syncthreads();
  }

  // ---- epilogue ----
  const float invT = 1.f / 250.f;
  for (int j = 0; j < 2; ++j)
    for (int i = 0; i < 8; ++i) {
      const size_t m = (size_t)(rb + g8 + i);
      const int n = nb0 + j * 16 + cn;
      out_s1[m * 512 + n] = s1[j][i] * invT;
      out_s2[m * 512 + n] = s2[j][i] * invT;
    }
  if (tid < 16 * 20) out_acc[rb * 20 + tid] = accs[tid];
}

// ------------------------------ launcher -----------------------------------

extern "C" void kernel_launch(void* const* d_in, const int* in_sizes, int n_in,
                              void* d_out, int out_size, void* d_ws, size_t ws_size,
                              hipStream_t stream) {
  const float* input  = (const float*)d_in[0];
  const float* mask   = (const float*)d_in[1];
  const float* h1m0   = (const float*)d_in[2];
  const float* h2m0   = (const float*)d_in[3];
  const float* om0    = (const float*)d_in[4];
  const float* W_in   = (const float*)d_in[5];
  const float* b_in   = (const float*)d_in[6];
  const float* W_h11  = (const float*)d_in[7];
  const float* b_h11  = (const float*)d_in[8];
  const float* W_h12  = (const float*)d_in[9];
  const float* b_h12  = (const float*)d_in[10];
  const float* W_h22  = (const float*)d_in[11];
  const float* b_h22  = (const float*)d_in[12];
  const float* W_out  = (const float*)d_in[13];
  const float* b_o    = (const float*)d_in[14];
  const float* tau_adp_h1 = (const float*)d_in[15];
  const float* tau_adp_h2 = (const float*)d_in[16];
  const float* tau_m_h1   = (const float*)d_in[17];
  const float* tau_m_h2   = (const float*)d_in[18];
  const float* tau_m_o    = (const float*)d_in[19];

  char* ws = (char*)d_ws;                       // bf16 weight arena, 2.33 MB
  __bf16* Wb_in  = (__bf16*)(ws + 0);           // 512*704*2 = 720896
  __bf16* Wb_h11 = (__bf16*)(ws + 720896);      // 512*512*2 = 524288
  __bf16* Wb_h12 = (__bf16*)(ws + 1245184);
  __bf16* Wb_h22 = (__bf16*)(ws + 1769472);
  __bf16* Wb_out = (__bf16*)(ws + 2293760);     // 32*512*2

  float* out     = (float*)d_out;
  float* out_acc = out;                          // [256][20]
  float* out_s1  = out + 5120;                   // [256][512]
  float* out_s2  = out + 5120 + 131072;          // [256][512]
  float* anorm   = out + 5120 + 2 * 131072;      // scalar A_norm

  prep_zero<<<1, 1, 0, stream>>>(anorm);
  prep_win <<<(512 * 704 + 255) / 256, 256, 0, stream>>>(W_in, Wb_in);
  prep_sq  <<<(512 * 512 + 255) / 256, 256, 0, stream>>>(W_h11, mask,             Wb_h11, anorm);
  prep_sq  <<<(512 * 512 + 255) / 256, 256, 0, stream>>>(W_h12, nullptr,          Wb_h12, anorm);
  prep_sq  <<<(512 * 512 + 255) / 256, 256, 0, stream>>>(W_h22, mask + 512 * 512, Wb_h22, anorm);
  prep_wout<<<(32 * 512 + 255) / 256, 256, 0, stream>>>(W_out, Wb_out);

  srnn_kernel<<<16, 512, 0, stream>>>(input, h1m0, h2m0, om0,
      b_in, b_h11, b_h12, b_h22, b_o,
      tau_adp_h1, tau_adp_h2, tau_m_h1, tau_m_h2, tau_m_o,
      Wb_in, Wb_h11, Wb_h12, Wb_h22, Wb_out,
      out_acc, out_s1, out_s2);
}
